// MPL_48481590837593
// MI455X (gfx1250) — matrix operand order repaired and checked by reading, not compiled, not run on hardware
//
#include <hip/hip_runtime.h>
#include <math.h>

typedef __attribute__((ext_vector_type(16))) _Float16 v16h;
typedef __attribute__((ext_vector_type(8)))  _Float16 v8h;
typedef __attribute__((ext_vector_type(8)))  float    v8f;
typedef __attribute__((ext_vector_type(4)))  unsigned int v4u;
typedef __attribute__((ext_vector_type(8)))  int v8i;
typedef __attribute__((ext_vector_type(4)))  int v4i;

#define B_  32
#define S_  10
#define D_  4096
#define K_  8192
#define MROWS 32          // padded rows per batch (21 used)
#define M_  (B_ * MROWS)  // 1024

// LDS tile layout: 128 rows x 32 halves, padded to 40 halves (80 B) per row.
// TDM pad feature: insert 4 DWORDs (16 B) after every 16 DWORDs (64 B = one row).
#define RS_H 40
#define TILE_BYTES (128 * RS_H * 2)   // 10240
#define OFF_A0 0
#define OFF_B0 (TILE_BYTES)
#define OFF_A1 (2 * TILE_BYTES)
#define OFF_B1 (3 * TILE_BYTES)

// ---------------------------------------------------------------------------
// Kernel 1: build A (f16, M_ x D_ row-major) in workspace.
//   row = b*32 + r :  r in [0,9]  -> p[b,r,:] * q_last[b,:]
//                     r in [10,19]-> p[b,r-10,:]^2
//                     r == 20     -> q_last[b,:]^2
//                     r in [21,31]-> 0 (padding)
// ---------------------------------------------------------------------------
__global__ void prep_A(const float* __restrict__ p, const float* __restrict__ q,
                       _Float16* __restrict__ A) {
  const int row = blockIdx.x;            // 0..M_-1
  const int b = row >> 5;
  const int r = row & 31;
  const int sidx = (r < 10) ? r : ((r < 20) ? (r - 10) : 0);
  const float* qlast = q + ((size_t)b * S_ + (S_ - 1)) * D_;
  const float* prow  = p + ((size_t)b * S_ + sidx) * D_;
  _Float16* outrow = A + (size_t)row * D_;
  for (int d = threadIdx.x; d < D_; d += blockDim.x) {
    float v;
    if (r < 10) {
      v = prow[d] * qlast[d];
    } else if (r < 20) {
      float x = prow[d]; v = x * x;
    } else if (r == 20) {
      float x = qlast[d]; v = x * x;
    } else {
      v = 0.0f;
    }
    outrow[d] = (_Float16)v;
  }
}

// ---------------------------------------------------------------------------
// Kernel 2: W2h[k,d] = (f16)(W[k,d]^2)
// ---------------------------------------------------------------------------
__global__ void prep_W2(const float* __restrict__ W, _Float16* __restrict__ W2h) {
  const size_t total = (size_t)K_ * D_;
  const size_t stride = (size_t)gridDim.x * blockDim.x;
  for (size_t i = (size_t)blockIdx.x * blockDim.x + threadIdx.x; i < total; i += stride) {
    float w = W[i];
    W2h[i] = (_Float16)(w * w);
  }
}

// ---------------------------------------------------------------------------
// Kernel 3: WMMA GEMM  C[M_ x K_] = A * W2h^T  (f16 in, f32 acc)
// 256 threads = 8 wave32; C tile 128x128; wave grid 2(M) x 4(N),
// each wave 64x32 = 4x2 WMMA tiles.  A/B tiles staged in LDS by the
// Tensor Data Mover (double buffered), fragments read from LDS.
// ---------------------------------------------------------------------------
__global__ __launch_bounds__(256)
void gemm_wmma(const _Float16* __restrict__ A, const _Float16* __restrict__ Wh,
               float* __restrict__ out) {
  // 64 KB: [0,40KB) = 4 stage buffers during main loop; whole array reused
  // as the 128x128 f32 C tile for the epilogue (after a barrier).
  __shared__ alignas(128) unsigned char smem[65536];
  _Float16* stage = (_Float16*)smem;
  float*    Cs    = (float*)smem;

  const int tid  = threadIdx.x;
  const int lane = tid & 31;
  const int wave = tid >> 5;
  const int wm = wave >> 2;              // 0..1 -> m offset wm*64
  const int wn = wave & 3;               // 0..3 -> n offset wn*32
  const int lr = lane & 15;
  const int lh = lane >> 4;

  const int mBlock = blockIdx.y * 128;
  const int nBlock = blockIdx.x * 128;

  v8f zero = {};
  v8f acc[4][2];
  #pragma unroll
  for (int i = 0; i < 4; ++i)
    #pragma unroll
    for (int j = 0; j < 2; ++j) acc[i][j] = zero;

  const int NK = D_ / 32;

#if __has_builtin(__builtin_amdgcn_tensor_load_to_lds) && __has_builtin(__builtin_amdgcn_s_wait_tensorcnt)
  // -------- TDM path (CDNA5 Tensor Data Mover, double buffered) ----------
  // Shared group-1 descriptor: 2D tile, tile_dim0=32 elem (d), tile_dim1=128
  // rows, data_size=2B, tensor_dim0=4096, dim0 stride=4096 elem,
  // LDS pad 4 DWORDs every 16 DWORDs (row stride 80 B).
  const v8i g1 = {
      (1 << 16) | (1 << 20) | (3 << 22) | (3 << 25), // data_size=2B, pad 16dw->+4dw
      (D_ & 0xffff) << 16,                           // tensor_dim0 lo16
      0,                                             // td0 hi16 | td1 lo16
      0x10 | (32 << 16),                             // td1 hi16 (=1<<20) | tile_dim0=32
      128,                                           // tile_dim1
      D_,                                            // tensor_dim0_stride lo32
      0, 0
  };
  const v4i g2 = {0, 0, 0, 0};
  const v4i g3 = {0, 0, 0, 0};
  const v8i g4 = {0, 0, 0, 0, 0, 0, 0, 0};          // 6-arg form: extra group
  const unsigned ldsBase = (unsigned)(size_t)(void*)smem; // low 32 bits = LDS addr

  auto issueTile = [&](const _Float16* gptr, unsigned ldsOff) {
    unsigned long long ga = (unsigned long long)(size_t)gptr;
    v4u g0;
    g0[0] = 1u;                                            // count=1, user D#
    g0[1] = ldsBase + ldsOff;                              // lds_addr (bytes)
    g0[2] = (unsigned)ga;                                  // global_addr[31:0]
    g0[3] = (unsigned)((ga >> 32) & 0x1FFFFFFu) | (2u << 30); // addr[56:32]|type=2
    __builtin_amdgcn_tensor_load_to_lds(g0, g1, g2, g3, g4, 0);
  };

  if (wave == 0) {
    issueTile(A  + (size_t)mBlock * D_, OFF_A0);
    issueTile(Wh + (size_t)nBlock * D_, OFF_B0);
  }

  for (int step = 0; step < NK; ++step) {
    const int cur = step & 1;
    if (wave == 0) __builtin_amdgcn_s_wait_tensorcnt(0);   // cur buffer filled
    __syncthreads();                                       // visible to all waves
    if (wave == 0 && step + 1 < NK) {                      // prefetch next tile
      const int kk = (step + 1) * 32;
      issueTile(A  + (size_t)mBlock * D_ + kk, cur ? OFF_A0 : OFF_A1);
      issueTile(Wh + (size_t)nBlock * D_ + kk, cur ? OFF_B0 : OFF_B1);
    }
    const _Float16* As = stage + (cur ? OFF_A1 : OFF_A0) / 2;
    const _Float16* Bs = stage + (cur ? OFF_B1 : OFF_B0) / 2;

    v16h af[4];
    #pragma unroll
    for (int i = 0; i < 4; ++i) {
      const _Float16* rb = As + (wm * 64 + i * 16 + lr) * RS_H + lh * 8;
      v8h lo = *(const v8h*)(rb);
      v8h hi = *(const v8h*)(rb + 16);
      af[i] = __builtin_shufflevector(lo, hi, 0, 1, 2, 3, 4, 5, 6, 7,
                                              8, 9, 10, 11, 12, 13, 14, 15);
    }
    v16h bf[2];
    #pragma unroll
    for (int j = 0; j < 2; ++j) {
      const _Float16* rb = Bs + (wn * 32 + j * 16 + lr) * RS_H + lh * 8;
      v8h lo = *(const v8h*)(rb);
      v8h hi = *(const v8h*)(rb + 16);
      bf[j] = __builtin_shufflevector(lo, hi, 0, 1, 2, 3, 4, 5, 6, 7,
                                              8, 9, 10, 11, 12, 13, 14, 15);
    }
    #pragma unroll
    for (int i = 0; i < 4; ++i)
      #pragma unroll
      for (int j = 0; j < 2; ++j)
        acc[i][j] = __builtin_amdgcn_wmma_f32_16x16x32_f16(
            false, af[i], false, bf[j], (short)0, acc[i][j], false, false);
    // no trailing barrier needed: buf[cur] is only re-issued after the
    // *next* iteration's barrier, which orders all reads above before it.
  }
#else
  // -------- fallback: cooperative staging, single buffer ------------------
  _Float16* As = stage + OFF_A0 / 2;
  _Float16* Bs = stage + OFF_B0 / 2;
  for (int step = 0; step < NK; ++step) {
    const int kk = step * 32;
    __syncthreads();
    {
      const int row = tid >> 1, col = (tid & 1) * 16;
      *(v8h*)(As + row * RS_H + col)     = *(const v8h*)(A + (size_t)(mBlock + row) * D_ + kk + col);
      *(v8h*)(As + row * RS_H + col + 8) = *(const v8h*)(A + (size_t)(mBlock + row) * D_ + kk + col + 8);
      *(v8h*)(Bs + row * RS_H + col)     = *(const v8h*)(Wh + (size_t)(nBlock + row) * D_ + kk + col);
      *(v8h*)(Bs + row * RS_H + col + 8) = *(const v8h*)(Wh + (size_t)(nBlock + row) * D_ + kk + col + 8);
    }
    __syncthreads();
    v16h af[4];
    #pragma unroll
    for (int i = 0; i < 4; ++i) {
      const _Float16* rb = As + (wm * 64 + i * 16 + lr) * RS_H + lh * 8;
      v8h lo = *(const v8h*)(rb);
      v8h hi = *(const v8h*)(rb + 16);
      af[i] = __builtin_shufflevector(lo, hi, 0, 1, 2, 3, 4, 5, 6, 7,
                                              8, 9, 10, 11, 12, 13, 14, 15);
    }
    v16h bf[2];
    #pragma unroll
    for (int j = 0; j < 2; ++j) {
      const _Float16* rb = Bs + (wn * 32 + j * 16 + lr) * RS_H + lh * 8;
      v8h lo = *(const v8h*)(rb);
      v8h hi = *(const v8h*)(rb + 16);
      bf[j] = __builtin_shufflevector(lo, hi, 0, 1, 2, 3, 4, 5, 6, 7,
                                              8, 9, 10, 11, 12, 13, 14, 15);
    }
    #pragma unroll
    for (int i = 0; i < 4; ++i)
      #pragma unroll
      for (int j = 0; j < 2; ++j)
        acc[i][j] = __builtin_amdgcn_wmma_f32_16x16x32_f16(
            false, af[i], false, bf[j], (short)0, acc[i][j], false, false);
  }
#endif

  __syncthreads();   // all stage-buffer reads done; smem now reused as Cs

  // Spill accumulators (f32 16x16 C layout: VGPR v -> M = v + lh*8, N = lr)
  #pragma unroll
  for (int i = 0; i < 4; ++i)
    #pragma unroll
    for (int j = 0; j < 2; ++j)
      #pragma unroll
      for (int v = 0; v < 8; ++v) {
        int row = wm * 64 + i * 16 + lh * 8 + v;
        int col = wn * 32 + j * 16 + lr;
        Cs[row * 128 + col] = acc[i][j][v];
      }
  __syncthreads();

  // Epilogue: 4 batches x 128 n x 10 s outputs per block; out layout (b,k,s)
  for (int idx = tid; idx < 4 * 128 * S_; idx += 256) {
    int s  = idx % S_;
    int nl = (idx / S_) & 127;
    int bl = idx / (S_ * 128);
    float dot = Cs[(bl * 32 + s) * 128 + nl];
    float np  = sqrtf(Cs[(bl * 32 + 10 + s) * 128 + nl]);
    float nq  = sqrtf(Cs[(bl * 32 + 20) * 128 + nl]);
    int b = blockIdx.y * 4 + bl;
    int n = blockIdx.x * 128 + nl;
    out[((size_t)b * K_ + n) * S_ + s] = -dot / (np * nq * (float)D_);
  }
}

// ---------------------------------------------------------------------------
extern "C" void kernel_launch(void* const* d_in, const int* in_sizes, int n_in,
                              void* d_out, int out_size, void* d_ws, size_t ws_size,
                              hipStream_t stream) {
  (void)in_sizes; (void)n_in; (void)out_size; (void)ws_size;
  const float* p = (const float*)d_in[0];
  const float* q = (const float*)d_in[1];
  const float* W = (const float*)d_in[2];
  float* out = (float*)d_out;

  // workspace: [ A f16: M_*D_ ][ W2h f16: K_*D_ ]  (~76 MB)
  _Float16* A  = (_Float16*)d_ws;
  _Float16* Wh = (_Float16*)d_ws + (size_t)M_ * D_;

  prep_A <<<dim3(M_),    dim3(256), 0, stream>>>(p, q, A);
  prep_W2<<<dim3(32768), dim3(256), 0, stream>>>(W, Wh);
  gemm_wmma<<<dim3(K_ / 128, M_ / 128), dim3(256), 0, stream>>>(A, Wh, out);
}
